// MixtralLayer_40072044871691
// MI455X (gfx1250) — matrix operand order repaired and checked by reading, not compiled, not run on hardware
//
#include <hip/hip_runtime.h>
#include <math.h>

// ---------------- model constants ----------------
static constexpr int kS   = 2048;
static constexpr int kHid = 1024;
static constexpr int kNH  = 16;
static constexpr int kNKV = 4;
static constexpr int kHD  = 64;
static constexpr int kQO  = 1536;   // (NH + 2*NKV) * HD
static constexpr int kE   = 8;
static constexpr int kFFN = 2048;

// ---------------- WMMA bf16 plumbing (CDNA5, wave32) ----------------
typedef __bf16 bf16x8  __attribute__((ext_vector_type(8)));
typedef __bf16 bf16x16 __attribute__((ext_vector_type(16)));
typedef float  f32x8   __attribute__((ext_vector_type(8)));
typedef unsigned int u32x4 __attribute__((ext_vector_type(4)));
typedef int i32x4 __attribute__((ext_vector_type(4)));
typedef int i32x8 __attribute__((ext_vector_type(8)));

__device__ __forceinline__ bf16x16 load_a_frag(const __bf16* rowptr, int k0, int lane) {
  const int ko = (lane & 16) ? 16 : 0;
  bf16x8 lo = *(const bf16x8*)(rowptr + k0 + ko);
  bf16x8 hi = *(const bf16x8*)(rowptr + k0 + 8 + ko);
  bf16x16 a;
#pragma unroll
  for (int i = 0; i < 8; ++i) { a[i] = lo[i]; a[i + 8] = hi[i]; }
  return a;
}

__device__ __forceinline__ bf16x16 load_b_frag(const __bf16* colptr, int k0, int lane) {
  const int ko = (lane & 16) ? 16 : 0;
  return *(const bf16x16*)(colptr + k0 + ko);
}

__device__ __forceinline__ f32x8 wmma_bf16(bf16x16 a, bf16x16 b, f32x8 c) {
  return __builtin_amdgcn_wmma_f32_16x16x32_bf16(false, a, false, b, (short)0, c, false, false);
}

__device__ __forceinline__ float rowmax16(float v) {
#pragma unroll
  for (int off = 8; off; off >>= 1) v = fmaxf(v, __shfl_xor(v, off, 16));
  return v;
}
__device__ __forceinline__ float rowsum16(float v) {
#pragma unroll
  for (int off = 8; off; off >>= 1) v += __shfl_xor(v, off, 16);
  return v;
}

// ---------------- Tensor Data Mover (TDM) helpers ----------------
#if defined(__has_builtin)
#if __has_builtin(__builtin_amdgcn_tensor_load_to_lds)
#define ATH_HAVE_TDM 1
#endif
#endif

#if defined(__HIP_DEVICE_COMPILE__)
#if defined(ATH_HAVE_TDM)
#warning "ATH DEVICE PASS: tensor_load_to_lds builtin AVAILABLE (TDM path compiled)"
#else
#warning "ATH DEVICE PASS: tensor_load_to_lds builtin NOT available (fallback copy path)"
#endif
#endif

__device__ __forceinline__ unsigned lds_offset_u32(const void* p) {
  return (unsigned)(unsigned long long)p;  // low 32 bits of generic LDS ptr = LDS offset
}

// D# group0/group1 for a 2D tile of 2-byte elements (ISA 8.3/8.4).
__device__ __forceinline__ void tdm_desc_2d(const void* gaddr, unsigned lds_off,
                                            unsigned tensor_d0, unsigned tensor_d1,
                                            unsigned tile_d0, unsigned tile_d1,
                                            unsigned long long stride0,
                                            u32x4& g0, i32x8& g1, bool gather16) {
  const unsigned long long ga = (unsigned long long)gaddr;
  g0[0] = 1u | (gather16 ? (1u << 31) : 0u);            // count=1; gather_mode, 16b idx
  g0[1] = lds_off;                                      // lds_addr
  g0[2] = (unsigned)(ga & 0xFFFFFFFFu);                 // global_addr[31:0]
  g0[3] = (unsigned)((ga >> 32) & 0x1FFFFFFu) | (2u << 30);  // addr[56:32] | type=2
  g1[0] = (1 << 16);                                    // wg_mask=0, data_size=1 (2B)
  g1[1] = (int)((tensor_d0 & 0xFFFFu) << 16);
  g1[2] = (int)(((tensor_d0 >> 16) & 0xFFFFu) | ((tensor_d1 & 0xFFFFu) << 16));
  g1[3] = (int)(((tensor_d1 >> 16) & 0xFFFFu) | ((tile_d0 & 0xFFFFu) << 16));
  g1[4] = (int)(tile_d1 & 0xFFFFu);
  g1[5] = (int)(stride0 & 0xFFFFFFFFull);
  g1[6] = (int)((stride0 >> 32) & 0xFFFFull);
  g1[7] = 0;
}

__device__ __forceinline__ void tdm_issue(u32x4 g0, i32x8 g1, i32x4 g2, i32x4 g3) {
#if defined(ATH_HAVE_TDM)
#if __clang_major__ >= 23
  i32x8 z = {};
  __builtin_amdgcn_tensor_load_to_lds(g0, g1, g2, g3, z, 0);
#else
  __builtin_amdgcn_tensor_load_to_lds(g0, g1, g2, g3, 0);
#endif
#else
  (void)g0; (void)g1; (void)g2; (void)g3;
#endif
}

__device__ __forceinline__ void tdm_wait() {
#if defined(ATH_HAVE_TDM)
  __builtin_amdgcn_s_wait_tensorcnt((short)0);
#endif
}

// ---------------- utility kernels ----------------
__global__ void k_f32_to_bf16(const float* __restrict__ src, __bf16* __restrict__ dst, int n) {
  for (int i = blockIdx.x * blockDim.x + threadIdx.x; i < n; i += gridDim.x * blockDim.x)
    dst[i] = (__bf16)src[i];
}

__global__ void k_zero_f32(float* p, int n) {
  for (int i = blockIdx.x * blockDim.x + threadIdx.x; i < n; i += gridDim.x * blockDim.x)
    p[i] = 0.0f;
}
__global__ void k_zero_i32(int* p, int n) {
  for (int i = blockIdx.x * blockDim.x + threadIdx.x; i < n; i += gridDim.x * blockDim.x)
    p[i] = 0;
}

__global__ void k_rmsnorm(const float* __restrict__ x, const float* __restrict__ w,
                          __bf16* __restrict__ out) {
  const int row = blockIdx.x, tid = threadIdx.x;
  const float* xr = x + (size_t)row * kHid;
  float ss = 0.0f;
#pragma unroll
  for (int j = 0; j < kHid / 256; ++j) { float v = xr[tid + j * 256]; ss += v * v; }
#pragma unroll
  for (int off = 16; off; off >>= 1) ss += __shfl_xor(ss, off, 32);
  __shared__ float wsums[8];
  if ((tid & 31) == 0) wsums[tid >> 5] = ss;
  __syncthreads();
  float tot = 0.0f;
#pragma unroll
  for (int i = 0; i < 8; ++i) tot += wsums[i];
  const float r = rsqrtf(tot * (1.0f / kHid) + 1e-5f);
#pragma unroll
  for (int j = 0; j < kHid / 256; ++j) {
    const int c = tid + j * 256;
    out[(size_t)row * kHid + c] = (__bf16)(xr[c] * r * w[c]);
  }
}

// ------ GEMM: qkv = h @ wqkv^T  (32-row M-block TDM-staged, B reused 2x) ----
__global__ void k_gemm_qkv(const __bf16* __restrict__ A, const __bf16* __restrict__ W,
                           float* __restrict__ C) {
  __shared__ alignas(16) __bf16 As[32 * kHid];   // 64KB
  const int lane = threadIdx.x & 31, wave = threadIdx.x >> 5;
  const int m0 = blockIdx.y * 32;
#if defined(ATH_HAVE_TDM)
  if (wave == 0) {
    u32x4 g0; i32x8 g1; i32x4 g2 = {}, g3 = {};
    tdm_desc_2d(A + (size_t)m0 * kHid, lds_offset_u32(As), kHid, 32, kHid, 32, kHid,
                g0, g1, false);
    tdm_issue(g0, g1, g2, g3);
    tdm_wait();
  }
#else
  for (int i = threadIdx.x; i < 32 * kHid; i += blockDim.x) As[i] = A[(size_t)m0 * kHid + i];
#endif
  __syncthreads();
  const int n0 = (blockIdx.x * 8 + wave) * 16;
  const __bf16* arow0 = As + (lane & 15) * kHid;
  const __bf16* arow1 = As + (16 + (lane & 15)) * kHid;
  const __bf16* bcol = W + (size_t)(n0 + (lane & 15)) * kHid;
  f32x8 c0 = {}, c1 = {};
  for (int k0 = 0; k0 < kHid; k0 += 32) {
    __builtin_prefetch(bcol + k0 + 256, 0, 1);
    const bf16x16 b = load_b_frag(bcol, k0, lane);
    c0 = wmma_bf16(load_a_frag(arow0, k0, lane), b, c0);
    c1 = wmma_bf16(load_a_frag(arow1, k0, lane), b, c1);
  }
#pragma unroll
  for (int r = 0; r < 8; ++r) {
    const int row = m0 + r + ((lane & 16) ? 8 : 0);
    C[(size_t)row * kQO + n0 + (lane & 15)] = c0[r];
    C[(size_t)(row + 16) * kQO + n0 + (lane & 15)] = c1[r];
  }
}

// ---------------- RoPE + split/scale/transpose ----------------
__global__ void k_rope(const float* __restrict__ qkv, __bf16* __restrict__ qb,
                       __bf16* __restrict__ kb, __bf16* __restrict__ vt) {
  const int s = blockIdx.x;
  const float* base = qkv + (size_t)s * kQO;
  for (int u = threadIdx.x; u < (kNH + kNKV) * 32; u += blockDim.x) {
    const int hh = u >> 5, i = u & 31;
    const float ang = (float)s * __expf(-9.2103403719761836f * (float)i / 32.0f);
    const float cs = __cosf(ang), sn = __sinf(ang);
    if (hh < kNH) {
      const float* p = base + hh * kHD;
      const float a = p[i], b = p[i + 32];
      __bf16* q = qb + ((size_t)s * kNH + hh) * kHD;
      q[i]      = (__bf16)(0.125f * (a * cs - b * sn));
      q[i + 32] = (__bf16)(0.125f * (b * cs + a * sn));
    } else {
      const int kv = hh - kNH;
      const float* p = base + kNH * kHD + kv * kHD;
      const float a = p[i], b = p[i + 32];
      __bf16* k = kb + ((size_t)s * kNKV + kv) * kHD;
      k[i]      = (__bf16)(a * cs - b * sn);
      k[i + 32] = (__bf16)(b * cs + a * sn);
    }
  }
  for (int u = threadIdx.x; u < kNKV * kHD; u += blockDim.x) {
    const int kv = u >> 6, d = u & 63;
    vt[((size_t)(kv * kHD + d)) * kS + s] =
        (__bf16)qkv[(size_t)s * kQO + (kNH + kNKV) * kHD + kv * kHD + d];
  }
}

// ---------------- flash attention: 1 wave per (head, 16-row q tile) --------
__global__ void k_attn(const __bf16* __restrict__ qb, const __bf16* __restrict__ kb,
                       const __bf16* __restrict__ vt, __bf16* __restrict__ attn) {
  __shared__ alignas(32) __bf16 pstage[4][16 * 32];
  const int lane = threadIdx.x & 31, wave = threadIdx.x >> 5;
  const int qtile = blockIdx.x * 4 + wave;
  const int head = blockIdx.y;
  const int kvh = head / (kNH / kNKV);
  const int qbase = qtile * 16, qlast = qbase + 15;

  const __bf16* qrow = qb + ((size_t)(qbase + (lane & 15)) * kNH + head) * kHD;
  bf16x16 qa[2];
  qa[0] = load_a_frag(qrow, 0, lane);
  qa[1] = load_a_frag(qrow, 32, lane);

  f32x8 o[4] = {};
  float m_r[8], l_r[8];
#pragma unroll
  for (int r = 0; r < 8; ++r) { m_r[r] = -3.0e30f; l_r[r] = 0.0f; }

  __bf16* ps = &pstage[wave][0];
  const int nchunk = qlast / 32 + 1;
  for (int c = 0; c < nchunk; ++c) {
    const int kb0 = c * 32;
    const bool do1 = (kb0 + 16) <= qlast;
    f32x8 s0 = {}, s1 = {};
    {
      const __bf16* kcol0 = kb + ((size_t)(kb0 + (lane & 15)) * kNKV + kvh) * kHD;
      s0 = wmma_bf16(qa[0], load_b_frag(kcol0, 0, lane), s0);
      s0 = wmma_bf16(qa[1], load_b_frag(kcol0, 32, lane), s0);
    }
    if (do1) {
      const __bf16* kcol1 = kb + ((size_t)(kb0 + 16 + (lane & 15)) * kNKV + kvh) * kHD;
      s1 = wmma_bf16(qa[0], load_b_frag(kcol1, 0, lane), s1);
      s1 = wmma_bf16(qa[1], load_b_frag(kcol1, 32, lane), s1);
    }
    float p0[8], p1[8];
#pragma unroll
    for (int r = 0; r < 8; ++r) {
      const int row = qbase + r + ((lane & 16) ? 8 : 0);
      const int key0 = kb0 + (lane & 15);
      float v0 = (key0 <= row) ? s0[r] : -3.0e30f;
      float v1 = (do1 && (key0 + 16 <= row)) ? s1[r] : -3.0e30f;
      const float mx = rowmax16(fmaxf(v0, v1));
      const float nm = fmaxf(m_r[r], mx);
      const float alpha = __expf(m_r[r] - nm);
      p0[r] = __expf(v0 - nm);
      p1[r] = __expf(v1 - nm);
      l_r[r] = l_r[r] * alpha + rowsum16(p0[r] + p1[r]);
      m_r[r] = nm;
#pragma unroll
      for (int b = 0; b < 4; ++b) o[b][r] *= alpha;
    }
#pragma unroll
    for (int r = 0; r < 8; ++r) {
      const int row = r + ((lane & 16) ? 8 : 0);
      ps[row * 32 + (lane & 15)]      = (__bf16)p0[r];
      ps[row * 32 + 16 + (lane & 15)] = (__bf16)p1[r];
    }
    asm volatile("s_wait_dscnt 0" ::: "memory");
    const bf16x16 pa = load_a_frag(ps + (lane & 15) * 32, 0, lane);
#pragma unroll
    for (int b = 0; b < 4; ++b) {
      const __bf16* vcol = vt + ((size_t)(kvh * kHD + b * 16 + (lane & 15))) * kS + kb0;
      o[b] = wmma_bf16(pa, load_b_frag(vcol, 0, lane), o[b]);
    }
  }
#pragma unroll
  for (int r = 0; r < 8; ++r) {
    const float inv = 1.0f / l_r[r];
    const int row = qbase + r + ((lane & 16) ? 8 : 0);
#pragma unroll
    for (int b = 0; b < 4; ++b)
      attn[(size_t)row * kHid + head * kHD + b * 16 + (lane & 15)] = (__bf16)(o[b][r] * inv);
  }
}

// ------ xres = x + attn @ wo^T (32-row M-block TDM-staged) ----------------
__global__ void k_gemm_wo(const __bf16* __restrict__ A, const __bf16* __restrict__ W,
                          const float* __restrict__ xin, float* __restrict__ xres) {
  __shared__ alignas(16) __bf16 As[32 * kHid];   // 64KB
  const int lane = threadIdx.x & 31, wave = threadIdx.x >> 5;
  const int m0 = blockIdx.y * 32;
#if defined(ATH_HAVE_TDM)
  if (wave == 0) {
    u32x4 g0; i32x8 g1; i32x4 g2 = {}, g3 = {};
    tdm_desc_2d(A + (size_t)m0 * kHid, lds_offset_u32(As), kHid, 32, kHid, 32, kHid,
                g0, g1, false);
    tdm_issue(g0, g1, g2, g3);
    tdm_wait();
  }
#else
  for (int i = threadIdx.x; i < 32 * kHid; i += blockDim.x) As[i] = A[(size_t)m0 * kHid + i];
#endif
  __syncthreads();
  const int n0 = (blockIdx.x * 8 + wave) * 16;
  const __bf16* arow0 = As + (lane & 15) * kHid;
  const __bf16* arow1 = As + (16 + (lane & 15)) * kHid;
  const __bf16* bcol = W + (size_t)(n0 + (lane & 15)) * kHid;
  f32x8 c0 = {}, c1 = {};
  for (int k0 = 0; k0 < kHid; k0 += 32) {
    __builtin_prefetch(bcol + k0 + 256, 0, 1);
    const bf16x16 b = load_b_frag(bcol, k0, lane);
    c0 = wmma_bf16(load_a_frag(arow0, k0, lane), b, c0);
    c1 = wmma_bf16(load_a_frag(arow1, k0, lane), b, c1);
  }
#pragma unroll
  for (int r = 0; r < 8; ++r) {
    const int row = m0 + r + ((lane & 16) ? 8 : 0);
    const size_t i0 = (size_t)row * kHid + n0 + (lane & 15);
    const size_t i1 = (size_t)(row + 16) * kHid + n0 + (lane & 15);
    xres[i0] = xin[i0] + c0[r];
    xres[i1] = xin[i1] + c1[r];
  }
}

// ---------------- MoE routing ----------------
__global__ void k_gate(const __bf16* __restrict__ h2, const float* __restrict__ gw,
                       float* __restrict__ logits) {
  const int idx = blockIdx.x * blockDim.x + threadIdx.x;
  if (idx >= kS * kE) return;
  const int t = idx >> 3, e = idx & 7;
  float s = 0.0f;
  for (int i = 0; i < kHid; ++i) s += (float)h2[(size_t)t * kHid + i] * gw[e * kHid + i];
  logits[idx] = s;
}

__global__ void k_route(const float* __restrict__ logits, float* __restrict__ wfull,
                        int* __restrict__ cnt, int* __restrict__ lists) {
  const int t = blockIdx.x * blockDim.x + threadIdx.x;
  if (t >= kS) return;
  float l[kE], m = -3.0e30f;
#pragma unroll
  for (int e = 0; e < kE; ++e) { l[e] = logits[t * kE + e]; m = fmaxf(m, l[e]); }
  float p[kE], sum = 0.0f;
#pragma unroll
  for (int e = 0; e < kE; ++e) { p[e] = __expf(l[e] - m); sum += p[e]; }
  int i0 = 0;
#pragma unroll
  for (int e = 1; e < kE; ++e) if (p[e] > p[i0]) i0 = e;
  int i1 = (i0 == 0) ? 1 : 0;
#pragma unroll
  for (int e = 0; e < kE; ++e) if (e != i0 && p[e] > p[i1]) i1 = e;
  const float denom = p[i0] + p[i1];
  wfull[t * kE + i0] = p[i0] / denom;
  wfull[t * kE + i1] = p[i1] / denom;
  int pos = atomicAdd(&cnt[i0], 1);
  lists[i0 * kS + pos] = t;
  pos = atomicAdd(&cnt[i1], 1);
  lists[i1 * kS + pos] = t;
}

// -- MoE GEMM1 (32-row M-block via two TDM gathers; B w1/w3 reused 2x) ------
__global__ void k_moe1(const __bf16* __restrict__ h2, const __bf16* __restrict__ w13,
                       const int* __restrict__ cnt, const int* __restrict__ lists,
                       __bf16* __restrict__ act) {
  __shared__ alignas(16) __bf16 As[32 * kHid];   // 64KB
  const int lane = threadIdx.x & 31, wave = threadIdx.x >> 5;
  const int e = blockIdx.z;
  const int ce = cnt[e];
  const int m0 = blockIdx.y * 32;
  if (m0 >= ce) return;
#if defined(ATH_HAVE_TDM)
  if (wave == 0) {
#pragma unroll
    for (int half = 0; half < 2; ++half) {
      int idx[16];
#pragma unroll
      for (int j = 0; j < 16; ++j) {
        const int m = m0 + half * 16 + j;
        idx[j] = lists[e * kS + (m < ce ? m : ce - 1)];
      }
      i32x4 g2, g3;
#pragma unroll
      for (int w = 0; w < 4; ++w) {
        g2[w] = (idx[2 * w] & 0xFFFF) | (idx[2 * w + 1] << 16);
        g3[w] = (idx[8 + 2 * w] & 0xFFFF) | (idx[8 + 2 * w + 1] << 16);
      }
      u32x4 g0; i32x8 g1;
      tdm_desc_2d(h2, lds_offset_u32(As + half * 16 * kHid), kHid, kS, kHid, 16, kHid,
                  g0, g1, true);
      tdm_issue(g0, g1, g2, g3);
    }
    tdm_wait();
  }
#else
  for (int i = threadIdx.x; i < 32 * kHid; i += blockDim.x) {
    const int m = m0 + (i >> 10);
    const int tok = lists[e * kS + (m < ce ? m : ce - 1)];
    As[i] = h2[(size_t)tok * kHid + (i & (kHid - 1))];
  }
#endif
  __syncthreads();
  const int f0 = (blockIdx.x * 8 + wave) * 16;
  const __bf16* arow0 = As + (lane & 15) * kHid;
  const __bf16* arow1 = As + (16 + (lane & 15)) * kHid;
  const __bf16* b1 = w13 + ((size_t)e * 2 * kFFN + f0 + (lane & 15)) * kHid;
  const __bf16* b3 = b1 + (size_t)kFFN * kHid;
  f32x8 c1a = {}, c3a = {}, c1b = {}, c3b = {};
  for (int k0 = 0; k0 < kHid; k0 += 32) {
    __builtin_prefetch(b1 + k0 + 256, 0, 1);
    __builtin_prefetch(b3 + k0 + 256, 0, 1);
    const bf16x16 a0 = load_a_frag(arow0, k0, lane);
    const bf16x16 a1 = load_a_frag(arow1, k0, lane);
    const bf16x16 bb1 = load_b_frag(b1, k0, lane);
    const bf16x16 bb3 = load_b_frag(b3, k0, lane);
    c1a = wmma_bf16(a0, bb1, c1a);
    c3a = wmma_bf16(a0, bb3, c3a);
    c1b = wmma_bf16(a1, bb1, c1b);
    c3b = wmma_bf16(a1, bb3, c3b);
  }
#pragma unroll
  for (int r = 0; r < 8; ++r) {
    const int ma = m0 + r + ((lane & 16) ? 8 : 0);
    const int mb = ma + 16;
    if (ma < ce) {
      const float y1 = c1a[r], y3 = c3a[r];
      act[((size_t)e * kS + ma) * kFFN + f0 + (lane & 15)] =
          (__bf16)((y1 / (1.0f + __expf(-y1))) * y3);
    }
    if (mb < ce) {
      const float y1 = c1b[r], y3 = c3b[r];
      act[((size_t)e * kS + mb) * kFFN + f0 + (lane & 15)] =
          (__bf16)((y1 / (1.0f + __expf(-y1))) * y3);
    }
  }
}

// -- MoE GEMM2 (32-row M-block, K staged in two 32x1024 TDM phases) ---------
__global__ void k_moe2(const __bf16* __restrict__ act, const __bf16* __restrict__ w2,
                       const int* __restrict__ cnt, const int* __restrict__ lists,
                       const float* __restrict__ wfull, float* __restrict__ moeout) {
  __shared__ alignas(16) __bf16 As[32 * 1024];   // 64KB: 32 rows x 1024-K chunk
  const int lane = threadIdx.x & 31, wave = threadIdx.x >> 5;
  const int e = blockIdx.z;
  const int ce = cnt[e];
  const int m0 = blockIdx.y * 32;
  if (m0 >= ce) return;
  const __bf16* atile = act + ((size_t)e * kS + m0) * kFFN;
  const int h0 = (blockIdx.x * 8 + wave) * 16;
  const __bf16* bcol = w2 + ((size_t)e * kHid + h0 + (lane & 15)) * kFFN;
  f32x8 c0 = {}, c1 = {};
#pragma unroll
  for (int kc = 0; kc < 2; ++kc) {
    __syncthreads();   // protect As reuse between phases
#if defined(ATH_HAVE_TDM)
    if (wave == 0) {
      u32x4 g0; i32x8 g1; i32x4 g2 = {}, g3 = {};
      tdm_desc_2d(atile + kc * 1024, lds_offset_u32(As), kFFN, 32, 1024, 32, kFFN,
                  g0, g1, false);
      tdm_issue(g0, g1, g2, g3);
      tdm_wait();
    }
#else
    for (int i = threadIdx.x; i < 32 * 1024; i += blockDim.x)
      As[i] = atile[(size_t)(i >> 10) * kFFN + kc * 1024 + (i & 1023)];
#endif
    __syncthreads();
    const __bf16* arow0 = As + (lane & 15) * 1024;
    const __bf16* arow1 = As + (16 + (lane & 15)) * 1024;
    for (int k0 = 0; k0 < 1024; k0 += 32) {
      __builtin_prefetch(bcol + kc * 1024 + k0 + 256, 0, 1);
      const bf16x16 b = load_b_frag(bcol, kc * 1024 + k0, lane);
      c0 = wmma_bf16(load_a_frag(arow0, k0, lane), b, c0);
      c1 = wmma_bf16(load_a_frag(arow1, k0, lane), b, c1);
    }
  }
#pragma unroll
  for (int r = 0; r < 8; ++r) {
    const int ma = m0 + r + ((lane & 16) ? 8 : 0);
    const int mb = ma + 16;
    if (ma < ce) {
      const int t = lists[e * kS + ma];
      atomicAdd(&moeout[(size_t)t * kHid + h0 + (lane & 15)], c0[r] * wfull[t * kE + e]);
    }
    if (mb < ce) {
      const int t = lists[e * kS + mb];
      atomicAdd(&moeout[(size_t)t * kHid + h0 + (lane & 15)], c1[r] * wfull[t * kE + e]);
    }
  }
}

__global__ void k_final_add(const float* __restrict__ xres, const float* __restrict__ moeout,
                            float* __restrict__ out, int n) {
  for (int i = blockIdx.x * blockDim.x + threadIdx.x; i < n; i += gridDim.x * blockDim.x)
    out[i] = xres[i] + moeout[i];
}

// ---------------- host launch ----------------
extern "C" void kernel_launch(void* const* d_in, const int* in_sizes, int n_in,
                              void* d_out, int out_size, void* d_ws, size_t ws_size,
                              hipStream_t stream) {
  (void)in_sizes; (void)n_in; (void)out_size; (void)ws_size;
  const float* x     = (const float*)d_in[0];
  const float* ln1_w = (const float*)d_in[1];
  const float* ln2_w = (const float*)d_in[2];
  const float* wqkv  = (const float*)d_in[3];
  const float* wo    = (const float*)d_in[4];
  const float* gatew = (const float*)d_in[5];
  const float* w13   = (const float*)d_in[6];
  const float* w2    = (const float*)d_in[7];
  float* out = (float*)d_out;

  char* ws = (char*)d_ws;
  size_t off = 0;
  auto alloc = [&](size_t bytes) -> char* {
    char* p = ws + off;
    off = (off + bytes + 255) & ~(size_t)255;
    return p;
  };
  __bf16* h1b    = (__bf16*)alloc((size_t)kS * kHid * 2);
  __bf16* wqkvb  = (__bf16*)alloc((size_t)kQO * kHid * 2);
  __bf16* wob    = (__bf16*)alloc((size_t)kHid * kHid * 2);
  __bf16* w13b   = (__bf16*)alloc((size_t)kE * 2 * kFFN * kHid * 2);
  __bf16* w2b    = (__bf16*)alloc((size_t)kE * kHid * kFFN * 2);
  float*  qkv    = (float*) alloc((size_t)kS * kQO * 4);
  __bf16* qb     = (__bf16*)alloc((size_t)kS * kNH * kHD * 2);
  __bf16* kb     = (__bf16*)alloc((size_t)kS * kNKV * kHD * 2);
  __bf16* vt     = (__bf16*)alloc((size_t)kNKV * kHD * kS * 2);
  __bf16* attnb  = (__bf16*)alloc((size_t)kS * kHid * 2);
  float*  xres   = (float*) alloc((size_t)kS * kHid * 4);
  __bf16* h2b    = (__bf16*)alloc((size_t)kS * kHid * 2);
  float*  logits = (float*) alloc((size_t)kS * kE * 4);
  float*  wfull  = (float*) alloc((size_t)kS * kE * 4);
  int*    cnt    = (int*)   alloc((size_t)kE * 4);
  int*    lists  = (int*)   alloc((size_t)kE * kS * 4);
  __bf16* actb   = (__bf16*)alloc((size_t)kE * kS * kFFN * 2);
  float*  moeout = (float*) alloc((size_t)kS * kHid * 4);

  // 1) weights -> bf16 (~106MB; fits in 192MB L2 for the GEMM streams)
  k_f32_to_bf16<<<4096, 256, 0, stream>>>(wqkv, wqkvb, kQO * kHid);
  k_f32_to_bf16<<<4096, 256, 0, stream>>>(wo, wob, kHid * kHid);
  k_f32_to_bf16<<<8192, 256, 0, stream>>>(w13, w13b, kE * 2 * kFFN * kHid);
  k_f32_to_bf16<<<8192, 256, 0, stream>>>(w2, w2b, kE * kHid * kFFN);

  // 2) RMSNorm1 -> bf16 activations
  k_rmsnorm<<<kS, 256, 0, stream>>>(x, ln1_w, h1b);

  // 3) QKV projection (WMMA bf16, 32-row TDM-staged A)
  k_gemm_qkv<<<dim3(kQO / 128, kS / 32), 256, 0, stream>>>(h1b, wqkvb, qkv);

  // 4) RoPE + bf16 split (q scaled by hd^-0.5), V transposed for P@V
  k_rope<<<kS, 256, 0, stream>>>(qkv, qb, kb, vt);

  // 5) flash attention (WMMA QK^T and P@V, LDS repack)
  k_attn<<<dim3(kS / 64, kNH), 128, 0, stream>>>(qb, kb, vt, attnb);

  // 6) output projection + residual
  k_gemm_wo<<<dim3(kHid / 128, kS / 32), 256, 0, stream>>>(attnb, wob, x, xres);

  // 7) RMSNorm2
  k_rmsnorm<<<kS, 256, 0, stream>>>(xres, ln2_w, h2b);

  // 8) routing: gate logits -> top2 -> per-expert token lists
  k_zero_i32<<<1, 64, 0, stream>>>(cnt, kE);
  k_zero_f32<<<64, 256, 0, stream>>>(wfull, kS * kE);
  k_gate<<<(kS * kE + 255) / 256, 256, 0, stream>>>(h2b, gatew, logits);
  k_route<<<kS / 256, 256, 0, stream>>>(logits, wfull, cnt, lists);

  // 9) MoE expert GEMMs (routed; TDM gather for token rows)
  k_zero_f32<<<2048, 256, 0, stream>>>(moeout, kS * kHid);
  k_moe1<<<dim3(kFFN / 128, kS / 32, kE), 256, 0, stream>>>(h2b, w13b, cnt, lists, actb);
  k_moe2<<<dim3(kHid / 128, kS / 32, kE), 256, 0, stream>>>(actb, w2b, cnt, lists, wfull, moeout);

  // 10) final residual
  k_final_add<<<2048, 256, 0, stream>>>(xres, moeout, out, kS * kHid);
}
